// SubResidualGeoLossl_8289286881958
// MI455X (gfx1250) — matrix operand order repaired and checked by reading, not compiled, not run on hardware
//
#include <hip/hip_runtime.h>
#include <hip/hip_bf16.h>

typedef __attribute__((ext_vector_type(16))) __bf16 v16bf;
typedef __attribute__((ext_vector_type(8)))  __bf16 v8bf;
typedef __attribute__((ext_vector_type(8)))  float  v8f;

#define IN_CH   64
#define CAT_CH  128   // 2*IN_CH
#define OUT_CH  64
#define KVOL    27
#define BOUND   2.0f

// ---------------------------------------------------------------------------
// Pack concat(x,y) -> bf16 feature table with zero sentinel row (row n).
// ---------------------------------------------------------------------------
__global__ void pack_feats_kernel(const float* __restrict__ x,
                                  const float* __restrict__ y,
                                  __bf16* __restrict__ feats, int n) {
    long i = (long)blockIdx.x * blockDim.x + threadIdx.x;
    long total = (long)(n + 1) * CAT_CH;
    if (i >= total) return;
    int row = (int)(i >> 7);
    int c   = (int)(i & (CAT_CH - 1));
    float v = 0.0f;
    if (row < n) v = (c < IN_CH) ? x[(long)row * IN_CH + c]
                                 : y[(long)row * IN_CH + (c - IN_CH)];
    feats[i] = (__bf16)v;
}

// Zero the sentinel row of the hidden bf16 table (row n, 64 channels).
__global__ void zero_h_sentinel_kernel(__bf16* __restrict__ h, int n) {
    int t = threadIdx.x;
    if (t < OUT_CH) h[(long)n * OUT_CH + t] = (__bf16)0.0f;
}

// ---------------------------------------------------------------------------
// Pre-swizzle weights into WMMA B-fragment layout.
// Fragment block b = ((k*KSTEPS + kk)*4 + nt): 512 bf16, lane-major:
//   element (lane*16 + t)  <->  w[k][ kk*32 + (lane>>4)*16 + t ][ nt*16 + (lane&15) ]
// so each lane's 16 K-values are one contiguous 32-byte load.
// ---------------------------------------------------------------------------
__global__ void pack_w_kernel(const float* __restrict__ w,
                              __bf16* __restrict__ ws,
                              int cin, int ksteps, int total) {
    int i = blockIdx.x * blockDim.x + threadIdx.x;
    if (i >= total) return;
    int b    = i >> 9;           // 512-element fragment id
    int lane = (i >> 4) & 31;
    int t    = i & 15;
    int k    = b / (ksteps * 4);
    int rem  = b - k * (ksteps * 4);
    int kk   = rem >> 2;
    int nt   = rem & 3;
    int K    = kk * 32 + ((lane >> 4) << 4) + t;
    int Nc   = nt * 16 + (lane & 15);
    ws[i] = (__bf16)w[((long)k * cin + K) * OUT_CH + Nc];
}

// ---------------------------------------------------------------------------
// Gather-GEMM layer.
//  - 8 waves/block; each wave computes TWO 16-row tiles x 64 channels.
//  - Weight slice for stencil tap k (16KB L1 / 8KB L2) staged in LDS,
//    double buffered: global->regs for k+1 issued before the WMMA burst of k,
//    regs->LDS after it, one barrier per k.
//  - A fragments (gathered rows, bf16) preloaded per k as a load clause.
// ---------------------------------------------------------------------------
template <int CIN, bool RELU, bool CLIP, typename OUT_T>
__global__ __launch_bounds__(256) void conv_kernel(
    const __bf16* __restrict__ src, const __bf16* __restrict__ wsw,
    const float* __restrict__ bias, const int* __restrict__ nbr,
    OUT_T* __restrict__ dst, int n) {
    constexpr int KSTEPS      = CIN / 32;
    constexpr int SLICE_BYTES = KSTEPS * 4 * 1024;     // per-k B slice
    constexpr int CH          = SLICE_BYTES / (256 * 16); // 16B chunks/thread

    __shared__ __align__(16) unsigned char smem[2 * SLICE_BYTES];

    const int wave = threadIdx.x >> 5;
    const int lane = threadIdx.x & 31;
    const int m    = lane & 15;   // A row within tile / B,C column
    const int kh   = lane >> 4;   // K-half selector
    const int pairs = (n >> 4) >> 1;
    int pair = blockIdx.x * 8 + wave;
    if (pair >= pairs) pair = pairs - 1;  // duplicate tail work (same values)
    const int row0 = pair * 32;           // tile0: rows row0.., tile1: row0+16..

    v8f c[2][4];
#pragma unroll
    for (int nt = 0; nt < 4; ++nt) {
        float bv = bias[nt * 16 + m];
#pragma unroll
        for (int t = 0; t < 2; ++t)
#pragma unroll
            for (int e = 0; e < 8; ++e) c[t][nt][e] = bv;
    }

    const uint4* gw = (const uint4*)wsw;
    uint4 tmp[CH];

    // Prologue: stage slice 0 into buffer 0.
#pragma unroll
    for (int q = 0; q < CH; ++q)
        tmp[q] = gw[threadIdx.x + q * 256];
    {
        uint4* s = (uint4*)smem;
#pragma unroll
        for (int q = 0; q < CH; ++q) s[threadIdx.x + q * 256] = tmp[q];
    }
    __syncthreads();

    for (int k = 0; k < KVOL; ++k) {
        // --- gather A fragments for both tiles (issued as one clause) ---
        v16bf a[2][KSTEPS];
#pragma unroll
        for (int t = 0; t < 2; ++t) {
            int idx = nbr[(size_t)k * n + row0 + t * 16 + m];
            const __bf16* arow = src + (size_t)idx * CIN;
#pragma unroll
            for (int kk = 0; kk < KSTEPS; ++kk) {
                // 16-bit A 16x32 lane layout (matches packed B):
                //   elems 0..15 = K(kk*32 + kh*16 ..)
                a[t][kk] = *(const v16bf*)(arow + kk * 32 + kh * 16);
            }
        }
        if (k + 1 < KVOL)
            __builtin_prefetch(&nbr[(size_t)(k + 1) * n + row0 + m], 0, 1);

        // --- issue staging loads for slice k+1 (completes behind WMMA burst) ---
        if (k + 1 < KVOL) {
#pragma unroll
            for (int q = 0; q < CH; ++q)
                tmp[q] = gw[(size_t)(k + 1) * (SLICE_BYTES / 16) +
                            threadIdx.x + q * 256];
        }

        // --- WMMA burst: B fragments from LDS, each reused for both tiles ---
        const unsigned char* sb = smem + (k & 1) * SLICE_BYTES;
#pragma unroll
        for (int kk = 0; kk < KSTEPS; ++kk) {
#pragma unroll
            for (int nt = 0; nt < 4; ++nt) {
                v16bf b = *(const v16bf*)(sb + ((kk * 4 + nt) << 10) + lane * 32);
#pragma unroll
                for (int t = 0; t < 2; ++t)
                    c[t][nt] = __builtin_amdgcn_wmma_f32_16x16x32_bf16(
                        false, a[t][kk], false, b, (short)0, c[t][nt],
                        false, false);
            }
        }

        // --- publish slice k+1 into the other LDS buffer ---
        if (k + 1 < KVOL) {
            uint4* s = (uint4*)(smem + ((k + 1) & 1) * SLICE_BYTES);
#pragma unroll
            for (int q = 0; q < CH; ++q) s[threadIdx.x + q * 256] = tmp[q];
        }
        __syncthreads();
    }

    // C/D layout: VGPR r, lane -> row M = r + kh*8, col N = nt*16 + m
#pragma unroll
    for (int t = 0; t < 2; ++t)
#pragma unroll
        for (int nt = 0; nt < 4; ++nt)
#pragma unroll
            for (int r = 0; r < 8; ++r) {
                float v = c[t][nt][r];
                if (RELU) v = v > 0.0f ? v : 0.0f;
                if (CLIP) v = fminf(fmaxf(v, -BOUND), BOUND);
                size_t rowm = (size_t)row0 + t * 16 + r + kh * 8;
                dst[rowm * OUT_CH + nt * 16 + m] = (OUT_T)v;
            }
}

// ---------------------------------------------------------------------------
static inline size_t align256(size_t x) { return (x + 255) & ~(size_t)255; }

extern "C" void kernel_launch(void* const* d_in, const int* in_sizes, int n_in,
                              void* d_out, int out_size, void* d_ws, size_t ws_size,
                              hipStream_t stream) {
    const float* x_feat = (const float*)d_in[0];
    const float* y_feat = (const float*)d_in[1];
    const float* w1     = (const float*)d_in[2];
    const float* b1     = (const float*)d_in[3];
    const float* w2     = (const float*)d_in[4];
    const float* b2     = (const float*)d_in[5];
    const int*   nbr    = (const int*)d_in[6];

    const int n = in_sizes[0] / IN_CH;  // 400000

    // Workspace layout (bf16 tables + swizzled weights), ~155 MB.
    char* ws = (char*)d_ws;
    size_t off = 0;
    __bf16* feats = (__bf16*)(ws + off);
    off += align256((size_t)(n + 1) * CAT_CH * sizeof(__bf16));
    __bf16* h = (__bf16*)(ws + off);
    off += align256((size_t)(n + 1) * OUT_CH * sizeof(__bf16));
    __bf16* w1s = (__bf16*)(ws + off);
    const int w1s_elems = KVOL * (CAT_CH / 32) * 4 * 512;  // 221184
    off += align256((size_t)w1s_elems * sizeof(__bf16));
    __bf16* w2s = (__bf16*)(ws + off);
    const int w2s_elems = KVOL * (IN_CH / 32) * 4 * 512;   // 110592
    (void)ws_size;

    // --- pack phase ---
    {
        long total = (long)(n + 1) * CAT_CH;
        int blocks = (int)((total + 255) / 256);
        pack_feats_kernel<<<blocks, 256, 0, stream>>>(x_feat, y_feat, feats, n);
    }
    zero_h_sentinel_kernel<<<1, 64, 0, stream>>>(h, n);
    pack_w_kernel<<<(w1s_elems + 255) / 256, 256, 0, stream>>>(
        w1, w1s, CAT_CH, CAT_CH / 32, w1s_elems);
    pack_w_kernel<<<(w2s_elems + 255) / 256, 256, 0, stream>>>(
        w2, w2s, IN_CH, IN_CH / 32, w2s_elems);

    // --- gather-GEMM layers: 8 waves/block, 2 tiles (32 rows) per wave ---
    const int pairs  = (n / 16) / 2;        // 12500
    const int blocks = (pairs + 7) / 8;     // 1563
    conv_kernel<CAT_CH, true,  false, __bf16>
        <<<blocks, 256, 0, stream>>>(feats, w1s, b1, nbr, h, n);
    conv_kernel<IN_CH,  false, true,  float>
        <<<blocks, 256, 0, stream>>>(h, w2s, b2, nbr, (float*)d_out, n);
}
